// MultiHeadAttention_69346541961782
// MI455X (gfx1250) — matrix operand order repaired and checked by reading, not compiled, not run on hardware
//
#include <hip/hip_runtime.h>
#include <hip/hip_bf16.h>

typedef __bf16 bf16;
typedef __attribute__((ext_vector_type(16))) __bf16 v16bf;
typedef __attribute__((ext_vector_type(8)))  float   v8f;

#define WMMA_BF16(A, B, C) \
  __builtin_amdgcn_wmma_f32_16x16x32_bf16(false, (A), false, (B), (short)0, (C), false, false)

constexpr int kDim   = 768;
constexpr int kHeads = 16;
constexpr int kHd    = 48;   // head dim
constexpr int kHp    = 64;   // padded head dim (zero-filled 48..63)
constexpr int kBatch = 4;
constexpr int kSeq   = 2048;
constexpr int kTok   = kBatch * kSeq;           // 8192 rows
constexpr int kNB    = 4;                       // GEMM N register blocking (16x64 per wave)
constexpr int kQT    = 2;                       // attention: query tiles per wave (32 queries)
constexpr float kScale = 0.03608439182435161f;  // 1/sqrt(768)  (reference scales by full dim)

// ---------------- fragment loaders (ISA 7.12.2 layouts, wave32) ----------------
// A (16x32 bf16): lane<16 holds row=lane, K: [k0..k0+7] then [k0+16..k0+23];
//                 lane>=16 holds row=lane-16, K: [k0+8..15] then [k0+24..31].
__device__ __forceinline__ v16bf load_a_frag(const bf16* __restrict__ base, int ld,
                                             int row0, int k0, int lane) {
  int half = lane >> 4, r = lane & 15;
  const bf16* p = base + (size_t)(row0 + r) * ld + k0 + half * 8;
  v16bf a;
  ((uint4*)&a)[0] = *(const uint4*)(p);
  ((uint4*)&a)[1] = *(const uint4*)(p + 16);
  return a;
}

// B (32x16 bf16), source with element (k, n) at base[n*ld + k].
// lane<16: col=lane,  K=[k0..k0+7] then [k0+16..k0+23]; lane>=16: col=lane-16, K=[k0+8..15] then [k0+24..31].
__device__ __forceinline__ v16bf load_b_frag(const bf16* __restrict__ base, int ld,
                                             int n0, int k0, int lane) {
  int half = lane >> 4, c = lane & 15;
  const bf16* p = base + (size_t)(n0 + c) * ld + k0 + half * 8;
  v16bf b;
  ((uint4*)&b)[0] = *(const uint4*)(p);
  ((uint4*)&b)[1] = *(const uint4*)(p + 16);
  return b;
}

// ---------------- prep kernels ----------------
__global__ void cvt_f32_to_bf16(const float* __restrict__ in, bf16* __restrict__ out, int n) {
  int i = blockIdx.x * 256 + threadIdx.x;
  if (i < n) out[i] = (bf16)in[i];
}

// W [K][N] f32 row-major -> WT [N][K] bf16
__global__ void transpose_w(const float* __restrict__ W, bf16* __restrict__ WT) {
  int i = blockIdx.x * 256 + threadIdx.x;
  if (i >= kDim * kDim) return;
  int n = i / kDim, k = i % kDim;
  WT[i] = (bf16)W[(size_t)k * kDim + n];
}

// ---------------- QKV projection: one wave = 16x64 output strip ----------------
// grid: (kTok/16, kDim/(16*kNB), 3)   block: 32
__global__ void qkv_gemm(const bf16* __restrict__ xbf,
                         const bf16* __restrict__ WTq, const bf16* __restrict__ WTk,
                         const bf16* __restrict__ WTv,
                         const float* __restrict__ bQ, const float* __restrict__ bK,
                         const float* __restrict__ bV,
                         bf16* __restrict__ Qbuf, bf16* __restrict__ Kbuf,
                         bf16* __restrict__ Vt) {
  int lane = threadIdx.x;
  int row0 = blockIdx.x * 16;
  int n0   = blockIdx.y * (16 * kNB);
  int which = blockIdx.z;
  const bf16*  WT   = (which == 0) ? WTq : (which == 1) ? WTk : WTv;
  const float* bias = (which == 0) ? bQ  : (which == 1) ? bK  : bV;

  v8f acc[kNB] = {};
  for (int k0 = 0; k0 < kDim; k0 += 32) {
    v16bf a = load_a_frag(xbf, kDim, row0, k0, lane);
    if (k0 + 64 < kDim)  // prefetch next-next A chunk (global_prefetch_b8)
      __builtin_prefetch(xbf + (size_t)(row0 + (lane & 15)) * kDim + k0 + 64, 0, 1);
#pragma unroll
    for (int t = 0; t < kNB; ++t) {
      v16bf bfr = load_b_frag(WT, kDim, n0 + 16 * t, k0, lane);
      acc[t] = WMMA_BF16(a, bfr, acc[t]);
    }
  }

  int rbase = row0 + ((lane >> 4) * 8);
#pragma unroll
  for (int t = 0; t < kNB; ++t) {
    int col = n0 + 16 * t + (lane & 15);
    float bv = bias[col];
    int h = col / kHd, dd = col % kHd;
#pragma unroll
    for (int j = 0; j < 8; ++j) {
      int tok = rbase + j;
      int b = tok >> 11;            // / kSeq
      int tt = tok & (kSeq - 1);
      float v = acc[t][j] + bv;
      if (which == 2) {
        // V transposed per head: [B][H][48][N]
        Vt[((size_t)(b * kHeads + h) * kHd + dd) * kSeq + tt] = (bf16)v;
      } else {
        bf16* dst = (which == 0) ? Qbuf : Kbuf;
        // padded per head: [B][H][N][64], cols 48..63 pre-zeroed
        dst[((size_t)(b * kHeads + h) * kSeq + tt) * kHp + dd] = (bf16)v;
      }
    }
  }
}

// ---------------- fused attention (flash-style online softmax) ----------------
// One wave handles kQT*16 = 32 queries; K/V fragments shared across query tiles.
// grid: (kSeq/(16*kQT), kBatch*kHeads)   block: 32 (one wave)
__global__ void attn_fused(const bf16* __restrict__ Qbuf, const bf16* __restrict__ Kbuf,
                           const bf16* __restrict__ Vt, bf16* __restrict__ ctx) {
  __shared__ bf16 ldsP[kQT][16 * 32];   // P tiles (16 queries x 32 keys) for C->A relayout
  int lane = threadIdx.x;
  int q0 = blockIdx.x * (16 * kQT);
  int bh = blockIdx.y;
  const bf16* Qh = Qbuf + (size_t)bh * kSeq * kHp;
  const bf16* Kh = Kbuf + (size_t)bh * kSeq * kHp;
  const bf16* Vh = Vt   + (size_t)bh * kHd * kSeq;

  v16bf qa[kQT][2];
  v8f m_run[kQT], l_run[kQT], acc[kQT][3];
#pragma unroll
  for (int qi = 0; qi < kQT; ++qi) {
    qa[qi][0] = load_a_frag(Qh, kHp, q0 + qi * 16, 0,  lane);
    qa[qi][1] = load_a_frag(Qh, kHp, q0 + qi * 16, 32, lane);
#pragma unroll
    for (int j = 0; j < 8; ++j) {
      m_run[qi][j] = -1e30f; l_run[qi][j] = 0.f;
      acc[qi][0][j] = 0.f; acc[qi][1][j] = 0.f; acc[qi][2][j] = 0.f;
    }
  }

  for (int kt = 0; kt < kSeq; kt += 32) {
    // K fragments for 32 keys, shared by all query tiles
    v16bf kb00 = load_b_frag(Kh, kHp, kt,      0,  lane);
    v16bf kb01 = load_b_frag(Kh, kHp, kt,      32, lane);
    v16bf kb10 = load_b_frag(Kh, kHp, kt + 16, 0,  lane);
    v16bf kb11 = load_b_frag(Kh, kHp, kt + 16, 32, lane);

#pragma unroll
    for (int qi = 0; qi < kQT; ++qi) {
      v8f z = {};
      v8f s0 = WMMA_BF16(qa[qi][0], kb00, z);  s0 = WMMA_BF16(qa[qi][1], kb01, s0);
      v8f s1 = WMMA_BF16(qa[qi][0], kb10, z);  s1 = WMMA_BF16(qa[qi][1], kb11, s1);

      v8f rm;
#pragma unroll
      for (int j = 0; j < 8; ++j) {
        s0[j] *= kScale; s1[j] *= kScale;
        rm[j] = fmaxf(s0[j], s1[j]);
      }
      for (int mask = 1; mask < 16; mask <<= 1) {
#pragma unroll
        for (int j = 0; j < 8; ++j) rm[j] = fmaxf(rm[j], __shfl_xor(rm[j], mask, 32));
      }

      v8f m_new, alpha, p0, p1, rs;
#pragma unroll
      for (int j = 0; j < 8; ++j) {
        m_new[j] = fmaxf(m_run[qi][j], rm[j]);
        alpha[j] = __expf(m_run[qi][j] - m_new[j]);
        m_run[qi][j] = m_new[j];
        p0[j] = __expf(s0[j] - m_new[j]);
        p1[j] = __expf(s1[j] - m_new[j]);
        rs[j] = p0[j] + p1[j];
      }
      for (int mask = 1; mask < 16; mask <<= 1) {
#pragma unroll
        for (int j = 0; j < 8; ++j) rs[j] += __shfl_xor(rs[j], mask, 32);
      }
#pragma unroll
      for (int j = 0; j < 8; ++j) {
        l_run[qi][j] = l_run[qi][j] * alpha[j] + rs[j];
        acc[qi][0][j] *= alpha[j]; acc[qi][1][j] *= alpha[j]; acc[qi][2][j] *= alpha[j];
      }

      // stage P (C layout, f32) -> LDS as bf16 for A-layout reload
      int half = lane >> 4, c = lane & 15;
#pragma unroll
      for (int j = 0; j < 8; ++j) {
        int r = j + 8 * half;
        ldsP[qi][r * 32 + c]      = (bf16)p0[j];
        ldsP[qi][r * 32 + 16 + c] = (bf16)p1[j];
      }
    }
    __syncthreads();

    v16bf pa[kQT];
    {
      int half = lane >> 4, r = lane & 15;
#pragma unroll
      for (int qi = 0; qi < kQT; ++qi) {
        const bf16* p = &ldsP[qi][r * 32 + half * 8];
        ((uint4*)&pa[qi])[0] = *(const uint4*)(p);
        ((uint4*)&pa[qi])[1] = *(const uint4*)(p + 16);
      }
    }
    __syncthreads();

    // V fragments (32 keys x 48 dims in 3 tiles), shared by all query tiles
    v16bf vb0 = load_b_frag(Vh, kSeq, 0,  kt, lane);
    v16bf vb1 = load_b_frag(Vh, kSeq, 16, kt, lane);
    v16bf vb2 = load_b_frag(Vh, kSeq, 32, kt, lane);
#pragma unroll
    for (int qi = 0; qi < kQT; ++qi) {
      acc[qi][0] = WMMA_BF16(pa[qi], vb0, acc[qi][0]);
      acc[qi][1] = WMMA_BF16(pa[qi], vb1, acc[qi][1]);
      acc[qi][2] = WMMA_BF16(pa[qi], vb2, acc[qi][2]);
    }
  }

  // epilogue: normalize and store ctx [B][N][768] bf16
  int c = lane & 15;
  int b = bh / kHeads, h = bh % kHeads;
#pragma unroll
  for (int qi = 0; qi < kQT; ++qi) {
    int rbase = q0 + qi * 16 + (lane >> 4) * 8;
#pragma unroll
    for (int j = 0; j < 8; ++j) {
      float inv = 1.0f / l_run[qi][j];
      size_t o = ((size_t)b * kSeq + (rbase + j)) * kDim + h * kHd;
      ctx[o + c]      = (bf16)(acc[qi][0][j] * inv);
      ctx[o + 16 + c] = (bf16)(acc[qi][1][j] * inv);
      ctx[o + 32 + c] = (bf16)(acc[qi][2][j] * inv);
    }
  }
}

// ---------------- output projection: ctx(bf16) x WP + bP -> f32 out ----------------
// grid: (kTok/16, kDim/(16*kNB))   block: 32
__global__ void proj_gemm(const bf16* __restrict__ ctx, const bf16* __restrict__ WTp,
                          const float* __restrict__ bP, float* __restrict__ out) {
  int lane = threadIdx.x;
  int row0 = blockIdx.x * 16;
  int n0   = blockIdx.y * (16 * kNB);
  v8f acc[kNB] = {};
  for (int k0 = 0; k0 < kDim; k0 += 32) {
    v16bf a = load_a_frag(ctx, kDim, row0, k0, lane);
    if (k0 + 64 < kDim)
      __builtin_prefetch(ctx + (size_t)(row0 + (lane & 15)) * kDim + k0 + 64, 0, 1);
#pragma unroll
    for (int t = 0; t < kNB; ++t) {
      v16bf bfr = load_b_frag(WTp, kDim, n0 + 16 * t, k0, lane);
      acc[t] = WMMA_BF16(a, bfr, acc[t]);
    }
  }
  int rbase = row0 + (lane >> 4) * 8;
#pragma unroll
  for (int t = 0; t < kNB; ++t) {
    int col = n0 + 16 * t + (lane & 15);
    float bv = bP[col];
#pragma unroll
    for (int j = 0; j < 8; ++j)
      out[(size_t)(rbase + j) * kDim + col] = acc[t][j] + bv;
  }
}

// ---------------- host launch ----------------
extern "C" void kernel_launch(void* const* d_in, const int* in_sizes, int n_in,
                              void* d_out, int out_size, void* d_ws, size_t ws_size,
                              hipStream_t stream) {
  const float* x  = (const float*)d_in[0];
  const float* WQ = (const float*)d_in[1];
  const float* bQ = (const float*)d_in[2];
  const float* WK = (const float*)d_in[3];
  const float* bK = (const float*)d_in[4];
  const float* WV = (const float*)d_in[5];
  const float* bV = (const float*)d_in[6];
  const float* WP = (const float*)d_in[7];
  const float* bP = (const float*)d_in[8];
  float* out = (float*)d_out;

  char* ws = (char*)d_ws;
  const size_t nX   = (size_t)kTok * kDim;                 // 6,291,456
  const size_t szX  = nX * sizeof(bf16);                   // 12,582,912
  const size_t szW  = (size_t)kDim * kDim * sizeof(bf16);  // 1,179,648
  const size_t szQK = (size_t)kBatch * kHeads * kSeq * kHp * sizeof(bf16);  // 16,777,216
  const size_t szV  = (size_t)kBatch * kHeads * kHd * kSeq * sizeof(bf16);  // 12,582,912

  bf16* xbf  = (bf16*)(ws);
  bf16* WTq  = (bf16*)(ws + szX);
  bf16* WTk  = (bf16*)(ws + szX + szW);
  bf16* WTv  = (bf16*)(ws + szX + 2 * szW);
  bf16* WTp  = (bf16*)(ws + szX + 3 * szW);
  bf16* Qbuf = (bf16*)(ws + szX + 4 * szW);
  bf16* Kbuf = (bf16*)(ws + szX + 4 * szW + szQK);
  bf16* Vt   = (bf16*)(ws + szX + 4 * szW + 2 * szQK);
  bf16* ctx  = (bf16*)(ws + szX + 4 * szW + 2 * szQK + szV);

  // zero Q/K pads (head dim 48 -> 64)
  hipMemsetAsync(Qbuf, 0, szQK, stream);
  hipMemsetAsync(Kbuf, 0, szQK, stream);

  cvt_f32_to_bf16<<<(int)((nX + 255) / 256), 256, 0, stream>>>(x, xbf, (int)nX);
  int wblocks = (kDim * kDim + 255) / 256;
  transpose_w<<<wblocks, 256, 0, stream>>>(WQ, WTq);
  transpose_w<<<wblocks, 256, 0, stream>>>(WK, WTk);
  transpose_w<<<wblocks, 256, 0, stream>>>(WV, WTv);
  transpose_w<<<wblocks, 256, 0, stream>>>(WP, WTp);

  dim3 gQKV(kTok / 16, kDim / (16 * kNB), 3);
  qkv_gemm<<<gQKV, 32, 0, stream>>>(xbf, WTq, WTk, WTv, bQ, bK, bV, Qbuf, Kbuf, Vt);

  dim3 gAtt(kSeq / (16 * kQT), kBatch * kHeads);
  attn_fused<<<gAtt, 32, 0, stream>>>(Qbuf, Kbuf, Vt, ctx);

  dim3 gP(kTok / 16, kDim / (16 * kNB));
  proj_gemm<<<gP, 32, 0, stream>>>(ctx, WTp, bP, out);
}
